// Encoder_49950469653250
// MI455X (gfx1250) — compile-verified, matrix operand order checked by reading, not run on hardware
//
#include <hip/hip_runtime.h>
#include <hip/hip_bf16.h>

typedef __attribute__((ext_vector_type(16))) _Float16 v16h;
typedef __attribute__((ext_vector_type(8)))  float    v8f;
typedef __attribute__((ext_vector_type(8)))  _Float16 h8;   // 16B vector of f16
typedef __attribute__((ext_vector_type(4)))  float    f4;   // 16B vector of f32
typedef __attribute__((ext_vector_type(4)))  int      i4;   // 16B vector of i32

#define NB      8
#define NPTS    8192
#define NPOINT_ 512
#define NSAMP   128
#define RAD2    0.04f          // 0.2^2
#define NROWS   524288         // 8*512*128
#define NQ      4096           // 8*512

// ---- gfx1250 async global->LDS copy ----
// Probe-confirmed: param 0 is `int4 __device__*` (AS1). Cast accordingly.
#define ASG(p) ((__attribute__((address_space(1))) i4*)(p))
#define ASL(p) ((__attribute__((address_space(3))) i4*)(p))
#if defined(__gfx1250__) && __has_builtin(__builtin_amdgcn_global_load_async_to_lds_b128)
#define HAVE_ASYNC_LDS 1
#endif

__device__ inline void wait_async0() {
#if defined(HAVE_ASYNC_LDS)
#if __has_builtin(__builtin_amdgcn_s_wait_asynccnt)
    __builtin_amdgcn_s_wait_asynccnt(0);
#else
    asm volatile("s_wait_asynccnt 0x0" ::: "memory");
#endif
#endif
}

// ---------------------------------------------------------------- FPS
// One block per batch. dist[] lives in LDS. Emits idx sequence matching
// jax.lax.scan semantics: slot 0 = point 0, slot i = argmax after i updates.
__global__ void fps_kernel(const float* __restrict__ x, float* __restrict__ nxyz)
{
    const int b   = blockIdx.x;
    const int tid = threadIdx.x;                 // 512 threads
    const float* xb = x + (size_t)b * 3 * NPTS;
    __shared__ float dist[NPTS];
    __shared__ float sv[512];
    __shared__ int   si[512];
    __shared__ int   s_cur;
    for (int n = tid; n < NPTS; n += 512) dist[n] = 1e10f;
    if (tid == 0) s_cur = 0;
    __syncthreads();
    for (int i = 0; i < NPOINT_; ++i) {
        const int cur = s_cur;
        const float px = xb[cur], py = xb[NPTS + cur], pz = xb[2*NPTS + cur];
        if (tid == 0) {
            float* o = nxyz + ((size_t)b * NPOINT_ + i) * 3;
            o[0] = px; o[1] = py; o[2] = pz;
        }
        float bv = -1.0f; int bi = 0;
        for (int n = tid; n < NPTS; n += 512) {
            float dx = xb[n] - px, dy = xb[NPTS + n] - py, dz = xb[2*NPTS + n] - pz;
            float d  = dx*dx + dy*dy + dz*dz;
            float dn = fminf(dist[n], d);
            dist[n] = dn;
            if (dn > bv) { bv = dn; bi = n; }    // strict > keeps lowest idx in-thread
        }
        sv[tid] = bv; si[tid] = bi;
        __syncthreads();
        for (int s = 256; s > 0; s >>= 1) {
            if (tid < s) {
                if (sv[tid+s] > sv[tid] || (sv[tid+s] == sv[tid] && si[tid+s] < si[tid])) {
                    sv[tid] = sv[tid+s]; si[tid] = si[tid+s];
                }
            }
            __syncthreads();
        }
        if (tid == 0) s_cur = si[0];
        __syncthreads();
    }
}

// ---------------------------------------------------------------- Ball query
// One wave32 per query point; ballot+prefix keeps the first <=128 indices in
// ascending order (== reference's sort-then-truncate), pad with first index.
__global__ void ballq_kernel(const float* __restrict__ x,
                             const float* __restrict__ nxyz,
                             int* __restrict__ bidx)
{
    const int tid  = threadIdx.x;
    const int lane = tid & 31;
    const int q    = blockIdx.x * 8 + (tid >> 5);
    const int b    = q >> 9;
    const float* xb = x + (size_t)b * 3 * NPTS;
    const float qx = nxyz[(size_t)q*3 + 0];
    const float qy = nxyz[(size_t)q*3 + 1];
    const float qz = nxyz[(size_t)q*3 + 2];
    int* out = bidx + (size_t)q * NSAMP;
    int count = 0, first = 0;
    for (int base = 0; base < NPTS && count < NSAMP; base += 32) {
        int n = base + lane;
        float dx = xb[n] - qx, dy = xb[NPTS + n] - qy, dz = xb[2*NPTS + n] - qz;
        bool pred = (dx*dx + dy*dy + dz*dz) <= RAD2;
        unsigned mask = __builtin_amdgcn_ballot_w32(pred);
        if (count == 0 && mask) first = base + __builtin_ctz(mask);
        int slot = count + __builtin_popcount(mask & ((1u << lane) - 1u));
        if (pred && slot < NSAMP) out[slot] = n;
        count += __builtin_popcount(mask);
    }
    if (count > NSAMP) count = NSAMP;
    for (int j = count + lane; j < NSAMP; j += 32) out[j] = first;
}

// ---------------------------------------------------------------- Layer 1 (K=3)
// Gather + tiny GEMM + BN1 partial sums. h1 stored f16 (bias included).
__global__ void layer1_kernel(const float* __restrict__ x,
                              const float* __restrict__ nxyz,
                              const int*   __restrict__ bidx,
                              const float* __restrict__ W1,
                              const float* __restrict__ b1,
                              _Float16* __restrict__ h1,
                              float* __restrict__ gSum, float* __restrict__ gSqs)
{
    const int q = blockIdx.x, tid = threadIdx.x;       // 128 threads
    const int b = q >> 9;
    __shared__ float Ws[64*3], Bsh[64], sS[64], sQ[64];
    if (tid < 64) {
        Ws[tid*3]   = W1[tid*3];
        Ws[tid*3+1] = W1[tid*3+1];
        Ws[tid*3+2] = W1[tid*3+2];
        Bsh[tid] = b1[tid]; sS[tid] = 0.f; sQ[tid] = 0.f;
    }
    __syncthreads();
    const float* xb = x + (size_t)b * 3 * NPTS;
    const int n = bidx[(size_t)q * NSAMP + tid];
    const float p0 = xb[n]           - nxyz[(size_t)q*3 + 0];
    const float p1 = xb[NPTS + n]    - nxyz[(size_t)q*3 + 1];
    const float p2 = xb[2*NPTS + n]  - nxyz[(size_t)q*3 + 2];
    _Float16* row = h1 + ((size_t)q * NSAMP + tid) * 64;
    #pragma unroll 8
    for (int o = 0; o < 64; ++o) {
        float v = fmaf(Ws[o*3], p0, fmaf(Ws[o*3+1], p1, fmaf(Ws[o*3+2], p2, Bsh[o])));
        row[o] = (_Float16)v;
        atomicAdd(&sS[o], v);
        atomicAdd(&sQ[o], v * v);
    }
    __syncthreads();
    if (tid < 64) { atomicAdd(&gSum[tid], sS[tid]); atomicAdd(&gSqs[tid], sQ[tid]); }
}

// ---------------------------------------------------------------- BN finalize
// y = g*(h-mu)/sqrt(var+eps)+be  ==  a*h + c
__global__ void bn_finalize_kernel(const float* __restrict__ gs, const float* __restrict__ gq,
                                   const float* __restrict__ g,  const float* __restrict__ be,
                                   float invcnt, float* __restrict__ aOut, float* __restrict__ cOut,
                                   int nch)
{
    int i = blockIdx.x * blockDim.x + threadIdx.x;
    if (i < nch) {
        float mu  = gs[i] * invcnt;
        float var = gq[i] * invcnt - mu * mu;
        float a   = g[i] * rsqrtf(var + 1e-5f);
        aOut[i] = a;
        cOut[i] = be[i] - mu * a;
    }
}

// ---------------------------------------------------------------- weight cvt
__global__ void cvt_f32_f16(const float* __restrict__ s, _Float16* __restrict__ d, int n)
{
    int i = blockIdx.x * 256 + threadIdx.x;
    if (i < n) d[i] = (_Float16)s[i];
}
__global__ void split_wh1_kernel(const float* __restrict__ Wh1,
                                 _Float16* __restrict__ Wm, float* __restrict__ Wx)
{
    int o = blockIdx.x;
    for (int k = threadIdx.x; k < 512; k += 256)
        Wm[(size_t)o*512 + k] = (_Float16)Wh1[(size_t)o*515 + k];
    if (threadIdx.x < 3)
        Wx[o*3 + threadIdx.x] = Wh1[(size_t)o*515 + 512 + threadIdx.x];
}

// ---------------------------------------------------------------- WMMA GEMM
// 128x128 block tile, BK=32, 256 threads = 8 waves; each wave: 2x4 tiles of
// v_wmma_f32_16x16x32_f16. A staged ->f16 with optional fused BN+ReLU.
// EPI 0: write raw f16 + per-channel sum/sumsq   (layers 2,3)
// EPI 1: +xyz aux-K (3 cols) + bias, leaky relu  (head 1, K=515)
// EPI 2: VAE split: mean | std=exp(0.5*clip(lv)) (head 2)
template<int AHALF, int BNORM, int EPI>
__global__ void gemm_kernel(const void* __restrict__ Aptr, int lda,
                            const float* __restrict__ bnA, const float* __restrict__ bnC,
                            const _Float16* __restrict__ Bw, int ldb,
                            const float* __restrict__ bias,
                            void* __restrict__ Outp, int ldc, int K, int Ntot,
                            const float* __restrict__ Xaux, const float* __restrict__ Waux,
                            float* __restrict__ gSum, float* __restrict__ gSqs,
                            float* __restrict__ out2)
{
    (void)Ntot;
    const int tid  = threadIdx.x;
    const int lane = tid & 31;
    const int w    = tid >> 5;
    const int wm   = w & 3;          // 4 wave rows  -> 32 M each
    const int wn   = w >> 2;         // 2 wave cols  -> 64 N each
    const int m0   = blockIdx.x * 128;
    const int n0   = blockIdx.y * 128;

    __shared__ _Float16 As[128 * 40];   // BK=32, +8 pad (80B row stride, 16B aligned)
    __shared__ _Float16 Bs[128 * 40];
    __shared__ float sS[128], sQ[128];
    if (EPI == 0 && tid < 128) { sS[tid] = 0.f; sQ[tid] = 0.f; }

    v8f acc[2][4];
    #pragma unroll
    for (int i = 0; i < 2; ++i)
        #pragma unroll
        for (int j = 0; j < 4; ++j) acc[i][j] = {};

    const int srow = tid >> 1;           // 128 rows, 2 threads/row
    const int sseg = (tid & 1) * 16;     // 16 f16 = 32B per thread

    for (int kt = 0; kt < K; kt += 32) {
        { // ---- stage A (optionally BN+ReLU), convert to f16, 16B vectors ----
            size_t g = (size_t)(m0 + srow) * lda + kt + sseg;
            float tmp[16];
            if (AHALF) {
                h8 a0 = *(const h8*)((const _Float16*)Aptr + g);
                h8 a1 = *(const h8*)((const _Float16*)Aptr + g + 8);
                #pragma unroll
                for (int i = 0; i < 8; ++i) { tmp[i] = (float)a0[i]; tmp[8+i] = (float)a1[i]; }
            } else {
                const f4* gs = (const f4*)((const float*)Aptr + g);
                f4 f0 = gs[0], f1 = gs[1], f2 = gs[2], f3 = gs[3];
                #pragma unroll
                for (int i = 0; i < 4; ++i) {
                    tmp[i] = f0[i]; tmp[4+i] = f1[i]; tmp[8+i] = f2[i]; tmp[12+i] = f3[i];
                }
            }
            h8 r0, r1;
            #pragma unroll
            for (int i = 0; i < 16; ++i) {
                float v = tmp[i];
                if (BNORM) {
                    int kg = kt + sseg + i;
                    v = fmaxf(0.f, fmaf(bnA[kg], v, bnC[kg]));
                }
                if (i < 8) r0[i] = (_Float16)v; else r1[i-8] = (_Float16)v;
            }
            h8* dst = (h8*)&As[srow * 40 + sseg];
            dst[0] = r0; dst[1] = r1;
        }
        { // ---- stage B: weights already f16, N-major [n][k] ----
            const _Float16* gsrc = &Bw[(size_t)(n0 + srow) * ldb + kt + sseg];
            _Float16* ldst = &Bs[srow * 40 + sseg];
#if defined(HAVE_ASYNC_LDS)
            __builtin_amdgcn_global_load_async_to_lds_b128(ASG(gsrc),     ASL(ldst),     0, 0);
            __builtin_amdgcn_global_load_async_to_lds_b128(ASG(gsrc + 8), ASL(ldst + 8), 0, 0);
#else
            h8 b0 = *(const h8*)gsrc;
            h8 b1 = *(const h8*)(gsrc + 8);
            ((h8*)ldst)[0] = b0;
            ((h8*)ldst)[1] = b1;
#endif
        }
        wait_async0();
        __syncthreads();

        // fragment loads per 16-bit A 16x32 layout (ISA 7.12.2)
        const int r  = lane & 15;
        const int kb = (lane >> 4) << 3;      // 0 or 8
        v16h af[2], bf[4];
        #pragma unroll
        for (int mt = 0; mt < 2; ++mt) {
            const _Float16* ap = &As[(wm*32 + mt*16 + r) * 40];
            #pragma unroll
            for (int v = 0; v < 8; ++v) {
                int k = ((v & 4) << 2) + kb + ((v & 3) << 1);
                af[mt][2*v]   = ap[k];
                af[mt][2*v+1] = ap[k + 1];
            }
        }
        #pragma unroll
        for (int nt = 0; nt < 4; ++nt) {
            const _Float16* bp = &Bs[(wn*64 + nt*16 + r) * 40];
            #pragma unroll
            for (int v = 0; v < 8; ++v) {
                int k = ((v & 4) << 2) + kb + ((v & 3) << 1);
                bf[nt][2*v]   = bp[k];
                bf[nt][2*v+1] = bp[k + 1];
            }
        }
        #pragma unroll
        for (int mt = 0; mt < 2; ++mt)
            #pragma unroll
            for (int nt = 0; nt < 4; ++nt)
                acc[mt][nt] = __builtin_amdgcn_wmma_f32_16x16x32_f16(
                    false, af[mt], false, bf[nt], (short)0, acc[mt][nt], false, false);
        __syncthreads();
    }

    // epilogue: C/D layout: lane -> N = lane&15, M = i + 8*(lane>>4)
    #pragma unroll
    for (int mt = 0; mt < 2; ++mt) {
        #pragma unroll
        for (int nt = 0; nt < 4; ++nt) {
            const int nl = wn*64 + nt*16 + (lane & 15);
            const int n  = n0 + nl;
            const float bb = bias[n];
            const int mbase = m0 + wm*32 + mt*16 + ((lane >> 4) << 3);
            float lsum = 0.f, lsq = 0.f;
            #pragma unroll
            for (int i = 0; i < 8; ++i) {
                const int m = mbase + i;
                float v = acc[mt][nt][i] + bb;
                if (EPI == 0) {
                    ((_Float16*)Outp)[(size_t)m * ldc + n] = (_Float16)v;
                    lsum += v;
                    lsq  += v * v;
                } else if (EPI == 1) {
                    const float* xr = &Xaux[(size_t)m * 3];
                    const float* wr = &Waux[(size_t)n * 3];
                    v += xr[0]*wr[0] + xr[1]*wr[1] + xr[2]*wr[2];
                    ((float*)Outp)[(size_t)m * ldc + n] = (v > 0.f) ? v : 0.01f * v;
                } else {
                    if (n < 512) {
                        ((float*)Outp)[(size_t)m * 512 + n] = v;
                    } else {
                        float lv = fminf(100.f, fmaxf(-100.f, v));
                        out2[(size_t)m * 512 + (n - 512)] = expf(0.5f * lv);
                    }
                }
            }
            if (EPI == 0) {                      // 1 atomic pair per tile per lane
                atomicAdd(&sS[nl], lsum);
                atomicAdd(&sQ[nl], lsq);
            }
        }
    }
    if (EPI == 0) {
        __syncthreads();
        if (tid < 128) {
            atomicAdd(&gSum[n0 + tid], sS[tid]);
            atomicAdd(&gSqs[n0 + tid], sQ[tid]);
        }
    }
}

// ---------------------------------------------------------------- maxpool (BN3+ReLU fused)
// b128 loads: thread (rg, cb) covers 8 channels over rows j = rg, rg+4, ...
__global__ void maxpool_kernel(const _Float16* __restrict__ h3,
                               const float* __restrict__ a3, const float* __restrict__ c3,
                               float* __restrict__ npts)
{
    const int q   = blockIdx.x, tid = threadIdx.x;   // 256 threads
    const int cb  = (tid & 63) * 8;                  // channel block (8 ch)
    const int rg  = tid >> 6;                        // row group 0..3
    __shared__ float red[4][512];
    float av[8], cv[8], mx[8];
    #pragma unroll
    for (int e = 0; e < 8; ++e) { av[e] = a3[cb+e]; cv[e] = c3[cb+e]; mx[e] = 0.f; }
    const _Float16* base = h3 + (size_t)q * NSAMP * 512 + cb;
    for (int j = rg; j < NSAMP; j += 4) {
        h8 hv = *(const h8*)(base + (size_t)j * 512);
        #pragma unroll
        for (int e = 0; e < 8; ++e) {
            float v = (float)hv[e];
            mx[e] = fmaxf(mx[e], fmaxf(0.f, fmaf(av[e], v, cv[e])));
        }
    }
    #pragma unroll
    for (int e = 0; e < 8; ++e) red[rg][cb + e] = mx[e];
    __syncthreads();
    if (rg == 0) {
        #pragma unroll
        for (int e = 0; e < 8; ++e) {
            float m = fmaxf(fmaxf(red[0][cb+e], red[1][cb+e]),
                            fmaxf(red[2][cb+e], red[3][cb+e]));
            npts[(size_t)q * 512 + cb + e] = m;
        }
    }
}

// ---------------------------------------------------------------- launch
extern "C" void kernel_launch(void* const* d_in, const int* in_sizes, int n_in,
                              void* d_out, int out_size, void* d_ws, size_t ws_size,
                              hipStream_t stream)
{
    (void)in_sizes; (void)n_in; (void)out_size; (void)ws_size;
    const float* x   = (const float*)d_in[0];
    const float* W1  = (const float*)d_in[2];
    const float* b1  = (const float*)d_in[3];
    const float* g1  = (const float*)d_in[4];
    const float* be1 = (const float*)d_in[5];
    const float* W2  = (const float*)d_in[6];
    const float* b2  = (const float*)d_in[7];
    const float* g2  = (const float*)d_in[8];
    const float* be2 = (const float*)d_in[9];
    const float* W3  = (const float*)d_in[10];
    const float* b3  = (const float*)d_in[11];
    const float* g3  = (const float*)d_in[12];
    const float* be3 = (const float*)d_in[13];
    const float* Wh1 = (const float*)d_in[14];
    const float* bh1 = (const float*)d_in[15];
    const float* Wh2 = (const float*)d_in[16];
    const float* bh2 = (const float*)d_in[17];

    char* ws = (char*)d_ws;
    size_t off = 0;
    auto take = [&](size_t bytes) -> char* {
        char* p = ws + off;
        off = (off + bytes + 255) & ~(size_t)255;
        return p;
    };
    float*    nxyz = (float*)   take((size_t)NQ * 3 * 4);
    int*      bidx = (int*)     take((size_t)NQ * NSAMP * 4);
    _Float16* h1   = (_Float16*)take((size_t)NROWS * 64  * 2);
    _Float16* h2   = (_Float16*)take((size_t)NROWS * 128 * 2);
    _Float16* h3   = (_Float16*)take((size_t)NROWS * 512 * 2);
    float*    npts = (float*)   take((size_t)NQ * 512 * 4);
    float*    tbuf = (float*)   take((size_t)NQ * 512 * 4);
    _Float16* W2h  = (_Float16*)take(128 * 64 * 2);
    _Float16* W3h  = (_Float16*)take(512 * 128 * 2);
    _Float16* Wh1h = (_Float16*)take((size_t)512 * 512 * 2);
    float*    Wh1x = (float*)   take(512 * 3 * 4);
    _Float16* Wh2h = (_Float16*)take((size_t)1024 * 512 * 2);
    float*    stats= (float*)   take(4096 * 4);
    float* gS1 = stats;        float* gQ1 = stats + 64;
    float* gS2 = stats + 128;  float* gQ2 = stats + 256;
    float* gS3 = stats + 384;  float* gQ3 = stats + 896;   // sums end at 1408
    float* a1 = stats + 1408;  float* c1 = a1 + 64;
    float* a2 = c1 + 64;       float* c2 = a2 + 128;
    float* a3 = c2 + 128;      float* c3 = a3 + 512;

    (void)hipMemsetAsync(stats, 0, 1408 * sizeof(float), stream);

    cvt_f32_f16<<<(128*64   + 255)/256, 256, 0, stream>>>(W2,  W2h,  128*64);
    cvt_f32_f16<<<(512*128  + 255)/256, 256, 0, stream>>>(W3,  W3h,  512*128);
    cvt_f32_f16<<<(1024*512 + 255)/256, 256, 0, stream>>>(Wh2, Wh2h, 1024*512);
    split_wh1_kernel<<<512, 256, 0, stream>>>(Wh1, Wh1h, Wh1x);

    fps_kernel  <<<NB,  512, 0, stream>>>(x, nxyz);
    ballq_kernel<<<NQ/8, 256, 0, stream>>>(x, nxyz, bidx);
    layer1_kernel<<<NQ, 128, 0, stream>>>(x, nxyz, bidx, W1, b1, h1, gS1, gQ1);
    bn_finalize_kernel<<<1, 64, 0, stream>>>(gS1, gQ1, g1, be1, 1.f/NROWS, a1, c1, 64);

    gemm_kernel<1,1,0><<<dim3(NROWS/128, 1), 256, 0, stream>>>(
        h1, 64, a1, c1, W2h, 64, b2, h2, 128, 64, 128,
        nullptr, nullptr, gS2, gQ2, nullptr);
    bn_finalize_kernel<<<1, 128, 0, stream>>>(gS2, gQ2, g2, be2, 1.f/NROWS, a2, c2, 128);

    gemm_kernel<1,1,0><<<dim3(NROWS/128, 4), 256, 0, stream>>>(
        h2, 128, a2, c2, W3h, 128, b3, h3, 512, 128, 512,
        nullptr, nullptr, gS3, gQ3, nullptr);
    bn_finalize_kernel<<<1, 512, 0, stream>>>(gS3, gQ3, g3, be3, 1.f/NROWS, a3, c3, 512);

    maxpool_kernel<<<NQ, 256, 0, stream>>>(h3, a3, c3, npts);

    gemm_kernel<0,0,1><<<dim3(NQ/128, 4), 256, 0, stream>>>(
        npts, 512, nullptr, nullptr, Wh1h, 512, bh1, tbuf, 512, 512, 512,
        nxyz, Wh1x, nullptr, nullptr, nullptr);

    float* outf = (float*)d_out;
    gemm_kernel<0,0,2><<<dim3(NQ/128, 8), 256, 0, stream>>>(
        tbuf, 512, nullptr, nullptr, Wh2h, 512, bh2, outf, 512, 512, 1024,
        nullptr, nullptr, nullptr, nullptr, outf + (size_t)NQ * 512);
}
